// Attention_249108103513
// MI455X (gfx1250) — matrix operand order repaired and checked by reading, not compiled, not run on hardware
//
#include <hip/hip_runtime.h>

// ---------------------------------------------------------------------------
// Non-local attention block, fused flash-attention style for gfx1250 (MI455X).
//
// B=4, C=256, CK=32, N=4096.  ~44 GFLOP, mandatory HBM traffic ~32MB.
// - scores never hit memory: online softmax per 16-query-row wave
// - CK==32 == WMMA K: one v_wmma_f32_16x16x32_f16 per 16x16 score tile
// - 64-key outer step amortizes softmax VALU; V streamed by the Tensor Data
//   Mover (tensor_load_to_lds) in 32-key chunks, double-buffered in LDS,
//   synchronized with s_wait_tensorcnt -> DMA overlaps WMMA at occupancy 1.
// - all f16 intermediates (18MB) stay resident in the 192MB L2.
// ---------------------------------------------------------------------------

#define Bn 4
#define Cc 256
#define CKk 32
#define Nn 4096

typedef _Float16 half8 __attribute__((ext_vector_type(8)));
typedef _Float16 v16h  __attribute__((ext_vector_type(16)));
typedef float    v8f   __attribute__((ext_vector_type(8)));
typedef unsigned int u32x4 __attribute__((ext_vector_type(4)));
typedef int          i32x8 __attribute__((ext_vector_type(8)));
typedef int          i32x4 __attribute__((ext_vector_type(4)));

union H16 { v16h v; half8 h[2]; };

// A-fragment (16x32 f16, MxK): lane L row M=L%16; half h=L/16:
// VGPR0..3 = K = h*8+[0..7], VGPR4..7 = K = 16+h*8+[0..7].  Two 16B loads.
static __device__ inline v16h ld_a_frag(const _Float16* row, int h) {
    H16 u;
    u.h[0] = *(const half8*)(row + h * 8);
    u.h[1] = *(const half8*)(row + 16 + h * 8);
    return u.v;
}

static __device__ inline v8f wmma_f16(v16h a, v16h b, v8f c) {
    return __builtin_amdgcn_wmma_f32_16x16x32_f16(false, a, false, b,
                                                  (short)0, c, false, false);
}

// TDM: DMA one V chunk (2D tile: 256 rows x 32 f16, row stride Nn*2 bytes)
// from global vT[d][m] into LDS packed as [256][32] f16.  Tracked by TENSORcnt.
// This toolchain exposes the 6-arg builtin:
//   (u32x4 g0, i32x8 g1, i32x4 g2, i32x4 g3, i32x8 extra, i32 cpol)
static __device__ inline void tdm_load_vchunk(unsigned long long gaddr,
                                              unsigned int lds_addr) {
    u32x4 g0;
    g0.x = 1u;                                        // count=1, user D#
    g0.y = lds_addr;                                  // LDS byte address
    g0.z = (unsigned int)gaddr;                       // global_addr[31:0]
    g0.w = (unsigned int)((gaddr >> 32) & 0x01FFFFFFull) | (2u << 30); // type=2
    i32x8 g1;
    g1[0] = 0x00010000;   // wg_mask=0, data_size=1 (2B), no barrier/iter/pad
    g1[1] = 0x10000000;   // tensor_dim0 = 4096 (bits 63:48)
    g1[2] = 0x01000000;   // tensor_dim1 = 256  (bits 95:80)
    g1[3] = 0x00200000;   // tile_dim0 = 32     (bits 127:112)
    g1[4] = 0x00000100;   // tile_dim1 = 256, tile_dim2 = 0
    g1[5] = 0x00001000;   // tensor_dim0_stride = 4096 elements
    g1[6] = 0;            // tensor_dim1_stride unused (2D)
    g1[7] = 0;
    i32x4 gz4 = {0, 0, 0, 0};                         // groups 2/3: 2D tensor
    i32x8 gz8 = {0, 0, 0, 0, 0, 0, 0, 0};
    __builtin_amdgcn_tensor_load_to_lds(g0, g1, gz4, gz4, gz8, 0);
}

// --- kernel 1: pack w1(32x256), w2(32x256), w3(256x256) -> f16 [320x256] ----
__global__ void cvt_w_kernel(const float* __restrict__ w1,
                             const float* __restrict__ w2,
                             const float* __restrict__ w3,
                             _Float16* __restrict__ Wall) {
    int idx = blockIdx.x * 256 + threadIdx.x;     // < 320*256
    int c = idx & 255;
    int r = idx >> 8;
    float v = (r < 32) ? w1[r * 256 + c]
            : (r < 64) ? w2[(r - 32) * 256 + c]
                       : w3[(r - 64) * 256 + c];
    Wall[idx] = (_Float16)v;
}

// --- kernel 2: x [B,C,N] f32 -> xT [B,N,C] f16, 64x64 LDS tile transpose ----
__global__ void cvt_x_kernel(const float* __restrict__ x,
                             _Float16* __restrict__ xT) {
    __shared__ float tile[64][65];                // pad: conflict-free columns
    int bid = blockIdx.x;                         // b*256 + cg*64 + ng
    int ng = bid & 63;  bid >>= 6;
    int cg = bid & 3;   bid >>= 2;
    int b  = bid;
    int n0 = ng * 64, c0 = cg * 64;
    int tid  = threadIdx.x;
    int col  = tid & 63;
    int rowq = tid >> 6;                          // 4 rows per sweep
    const float* xb = x + (size_t)b * Cc * Nn;
    #pragma unroll
    for (int i = 0; i < 16; ++i) {                // coalesced 256B row reads
        int c = rowq + i * 4;
        tile[c][col] = xb[(size_t)(c0 + c) * Nn + n0 + col];
    }
    __syncthreads();
    _Float16* xTb = xT + (size_t)b * Nn * Cc;
    #pragma unroll
    for (int i = 0; i < 16; ++i) {                // coalesced 128B row writes
        int n = rowq + i * 4;
        xTb[(size_t)(n0 + n) * Cc + c0 + col] = (_Float16)tile[col][n];
    }
}

// --- kernel 3: projection GEMM  proj[320,N] = Wall[320,256] @ x[256,N] ------
// One wave: 16(M) x 64(N) strip, 32 WMMAs.  Rows 0-31 -> q[N,32],
// 32-63 -> k[N,32], 64-319 -> vT[256,N] (natural D-layout store).
__global__ void __launch_bounds__(32)
proj_kernel(const _Float16* __restrict__ Wall,
            const _Float16* __restrict__ xT,
            _Float16* __restrict__ q,
            _Float16* __restrict__ k,
            _Float16* __restrict__ vT) {
    int wid = blockIdx.x;             // b*20*64 + mt*64 + ng
    int ng = wid & 63;  wid >>= 6;
    int mt = wid % 20;
    int b  = wid / 20;
    int lane = threadIdx.x;
    int h  = lane >> 4;
    int ln = lane & 15;
    int m0 = mt * 16;
    int n0 = ng * 64;

    const _Float16* xTb  = xT + (size_t)b * Nn * Cc;
    const _Float16* arow = Wall + (size_t)(m0 + ln) * Cc;

    v8f acc[4];
    #pragma unroll
    for (int j = 0; j < 4; ++j) { v8f z = {}; acc[j] = z; }

    #pragma unroll
    for (int cc = 0; cc < Cc; cc += 32) {
        v16h a = ld_a_frag(arow + cc, h);
        const _Float16* bbase = xTb + cc;            // B frag: same K order as A frag
        #pragma unroll
        for (int j = 0; j < 4; ++j) {
            v16h bf = ld_a_frag(bbase + (size_t)(n0 + j * 16 + ln) * Cc, h);
            acc[j] = wmma_f16(a, bf, acc[j]);
        }
    }

    _Float16* qb = q  + (size_t)b * Nn * CKk;
    _Float16* kb = k  + (size_t)b * Nn * CKk;
    _Float16* vb = vT + (size_t)b * Cc * Nn;
    #pragma unroll
    for (int j = 0; j < 4; ++j) {
        int n = n0 + j * 16 + ln;
        #pragma unroll
        for (int r = 0; r < 8; ++r) {
            int m = m0 + r + 8 * h;                 // D layout: M = r + 8*half
            _Float16 hv = (_Float16)acc[j][r];
            if (m0 < 32)       qb[(size_t)n * CKk + m] = hv;
            else if (m0 < 64)  kb[(size_t)n * CKk + (m - 32)] = hv;
            else               vb[(size_t)(m - 64) * Nn + n] = hv;
        }
    }
}

// --- kernel 4: fused flash attention, one wave per 16 query rows ------------
__global__ void __launch_bounds__(32)
attn_kernel(const _Float16* __restrict__ q,
            const _Float16* __restrict__ k,
            const _Float16* __restrict__ vT,
            const float* __restrict__ x,
            float* __restrict__ out) {
    __shared__ __align__(128) _Float16 v_lds[2][32 * 256]; // 2 x 16KB V chunks
    __shared__ _Float16 p_lds[16 * 64];                    // P staging (2KB)
    __shared__ float    o_lds[16][17];                     // epilogue transpose

    int wid = blockIdx.x;                 // b*256 + qt
    int qt = wid & 255;
    int b  = wid >> 8;
    int lane = threadIdx.x;
    int h  = lane >> 4;
    int ln = lane & 15;
    int n0 = qt * 16;

    const _Float16* qb = q  + (size_t)b * Nn * CKk;
    const _Float16* kb = k  + (size_t)b * Nn * CKk;
    const _Float16* vb = vT + (size_t)b * Cc * Nn;
    unsigned long long vg = (unsigned long long)(uintptr_t)vb; // byte addr
    unsigned int lbuf0 = (unsigned int)(uintptr_t)&v_lds[0][0];
    unsigned int lbuf1 = (unsigned int)(uintptr_t)&v_lds[1][0];

    // prologue: DMA V chunks 0 and 1 (32 keys x 256 d each)
    tdm_load_vchunk(vg, lbuf0);
    tdm_load_vchunk(vg + 64ull, lbuf1);

    // Q fragment loaded once: CK==32 == WMMA K.
    v16h qa = ld_a_frag(qb + (size_t)(n0 + ln) * CKk, h);

    float mrow[8], lrow[8], corr[8];
    v8f oacc[16];                          // 16 d-tiles x 8 f32 = full d=256
    #pragma unroll
    for (int r = 0; r < 8; ++r) { mrow[r] = -1e30f; lrow[r] = 0.0f; }
    #pragma unroll
    for (int t = 0; t < 16; ++t) { v8f z = {}; oacc[t] = z; }

    for (int step = 0; step < Nn / 64; ++step) {
        int mo = step * 64;
        // -- scores: s[16q x 64keys] via 4 WMMAs (K frags in A-frag K order) --
        v16h kb0 = ld_a_frag(kb + (size_t)(mo +      ln) * CKk, h);
        v16h kb1 = ld_a_frag(kb + (size_t)(mo + 16 + ln) * CKk, h);
        v16h kb2 = ld_a_frag(kb + (size_t)(mo + 32 + ln) * CKk, h);
        v16h kb3 = ld_a_frag(kb + (size_t)(mo + 48 + ln) * CKk, h);
        v8f z = {};
        v8f s0 = wmma_f16(qa, kb0, z);
        v8f s1 = wmma_f16(qa, kb1, z);
        v8f s2 = wmma_f16(qa, kb2, z);
        v8f s3 = wmma_f16(qa, kb3, z);

        // -- online softmax over 64 cols; row M=r+8h spans one 16-lane half --
        #pragma unroll
        for (int r = 0; r < 8; ++r) {
            float mx = fmaxf(fmaxf(s0[r], s1[r]), fmaxf(s2[r], s3[r]));
            #pragma unroll
            for (int off = 8; off; off >>= 1)
                mx = fmaxf(mx, __shfl_xor(mx, off, 32));
            float mnew = fmaxf(mrow[r], mx);
            float c  = __expf(mrow[r] - mnew);
            float e0 = __expf(s0[r] - mnew);
            float e1 = __expf(s1[r] - mnew);
            float e2 = __expf(s2[r] - mnew);
            float e3 = __expf(s3[r] - mnew);
            float ps = (e0 + e1) + (e2 + e3);
            #pragma unroll
            for (int off = 8; off; off >>= 1)
                ps += __shfl_xor(ps, off, 32);
            lrow[r] = lrow[r] * c + ps;
            mrow[r] = mnew;
            corr[r] = c;
            int row = r + 8 * h;
            p_lds[row * 64 +      ln] = (_Float16)e0;
            p_lds[row * 64 + 16 + ln] = (_Float16)e1;
            p_lds[row * 64 + 32 + ln] = (_Float16)e2;
            p_lds[row * 64 + 48 + ln] = (_Float16)e3;
        }
        __syncthreads();

        // rescale running output once per 64 keys
        #pragma unroll
        for (int t = 0; t < 16; ++t)
            #pragma unroll
            for (int r = 0; r < 8; ++r)
                oacc[t][r] *= corr[r];

        // -- o += P @ V over two 32-key chunks, V from TDM-filled LDS --------
        #pragma unroll
        for (int s = 0; s < 2; ++s) {
            int j = step * 2 + s;                  // global chunk index
            if (j < 127) __builtin_amdgcn_s_wait_tensorcnt(1);
            else         __builtin_amdgcn_s_wait_tensorcnt(0);
            asm volatile("" ::: "memory");         // keep ds reads below wait
            const _Float16* vbuf = (j & 1) ? &v_lds[1][0] : &v_lds[0][0];
            v16h pa = ld_a_frag(p_lds + ln * 64 + s * 32, h);
            #pragma unroll
            for (int t = 0; t < 16; ++t) {
                v16h vf = ld_a_frag(vbuf + (size_t)(t * 16 + ln) * 32, h);
                oacc[t] = wmma_f16(pa, vf, oacc[t]);
            }
            asm volatile("" ::: "memory");         // ds reads done before refill
            if (j + 2 < 128)
                tdm_load_vchunk(vg + (unsigned long long)(j + 2) * 64ull,
                                (j & 1) ? lbuf1 : lbuf0);
        }
        __syncthreads();                           // p_lds reuse next step
    }

    // -- epilogue: out[b,d,n] = o[n,d]/l[n] + x[b,d,n], coalesced via LDS ----
    float rl[8];
    #pragma unroll
    for (int r = 0; r < 8; ++r) rl[r] = 1.0f / lrow[r];
    const float* xb = x   + (size_t)b * Cc * Nn;
    float*       ob = out + (size_t)b * Cc * Nn;
    #pragma unroll
    for (int t = 0; t < 16; ++t) {
        __syncthreads();
        #pragma unroll
        for (int r = 0; r < 8; ++r)
            o_lds[r + 8 * h][ln] = oacc[t][r] * rl[r];   // [n_local][d_local]
        __syncthreads();
        #pragma unroll
        for (int dl = 0; dl < 8; ++dl) {
            int d = t * 16 + dl * 2 + h;
            size_t idx = (size_t)d * Nn + n0 + ln;       // 16 consecutive n
            ob[idx] = o_lds[ln][dl * 2 + h] + xb[idx];
        }
    }
}

// ---------------------------------------------------------------------------
extern "C" void kernel_launch(void* const* d_in, const int* in_sizes, int n_in,
                              void* d_out, int out_size, void* d_ws, size_t ws_size,
                              hipStream_t stream) {
    (void)in_sizes; (void)n_in; (void)out_size; (void)ws_size;
    const float* x  = (const float*)d_in[0];
    const float* w1 = (const float*)d_in[1];
    const float* w2 = (const float*)d_in[2];
    const float* w3 = (const float*)d_in[3];
    float* out = (float*)d_out;

    // f16 workspace: Wall 160KB | xT 8MB | q 1MB | k 1MB | vT 8MB  (~18.2MB)
    char* ws = (char*)d_ws;
    _Float16* Wall = (_Float16*)(ws);
    _Float16* xT   = (_Float16*)(ws + 163840);
    _Float16* qbuf = (_Float16*)(ws + 163840 + 8388608);
    _Float16* kbuf = (_Float16*)(ws + 163840 + 8388608 + 1048576);
    _Float16* vT   = (_Float16*)(ws + 163840 + 8388608 + 2097152);

    cvt_w_kernel<<<(320 * 256) / 256, 256, 0, stream>>>(w1, w2, w3, Wall);
    cvt_x_kernel<<<Bn * 4 * 64, 256, 0, stream>>>(x, xT);
    proj_kernel<<<Bn * 20 * 64, 32, 0, stream>>>(Wall, xT, qbuf, kbuf, vT);
    attn_kernel<<<Bn * 256, 32, 0, stream>>>(qbuf, kbuf, vT, x, out);
}
